// _RoutedExpertsTorchEP_78099685310887
// MI455X (gfx1250) — matrix operand order repaired and checked by reading, not compiled, not run on hardware
//
#include <hip/hip_runtime.h>
#include <hip/hip_bf16.h>
#include <cstdint>
#include <cstddef>

// Problem constants (from reference)
#define N_TOK     2048
#define D_MODEL   1024
#define D_INT     512
#define N_EXPERTS 8
#define TOP_K     2
#define N_SLOTS   (N_TOK * TOP_K)               // 4096 (token,slot) pairs
#define MTILE     32                            // slots per block (2 WMMA M-tiles)
#define MAX_LIST  (N_SLOTS + N_EXPERTS * MTILE) // padded slot list upper bound
#define MAX_TILES (MAX_LIST / MTILE)            // 136 tiles of 32 slots

typedef __attribute__((ext_vector_type(16))) __bf16 v16bf;
typedef __attribute__((ext_vector_type(8)))  float  v8f;
typedef __attribute__((ext_vector_type(4)))  __bf16 v4bf;
typedef __attribute__((address_space(3))) char lds_char_t;

union FragAB { uint4 q[2]; v16bf v; };

// ---------------------------------------------------------------------------
// f32 -> bf16 bulk conversion (4 elements / thread, b128 in, b64 out)
// ---------------------------------------------------------------------------
__global__ void cvt_f32_to_bf16(const float* __restrict__ src,
                                __bf16* __restrict__ dst, int n4) {
  int i = blockIdx.x * blockDim.x + threadIdx.x;
  if (i >= n4) return;
  float4 f = reinterpret_cast<const float4*>(src)[i];
  v4bf h;
  h.x = (__bf16)f.x; h.y = (__bf16)f.y; h.z = (__bf16)f.z; h.w = (__bf16)f.w;
  reinterpret_cast<v4bf*>(dst)[i] = h;
}

// ---------------------------------------------------------------------------
// Build per-expert slot lists (padded to multiples of MTILE) + tile metadata.
// ---------------------------------------------------------------------------
__global__ void build_dispatch(const int* __restrict__ indices,
                               const int* __restrict__ counts,
                               int* __restrict__ slot_list,
                               int* __restrict__ tile_meta,
                               int* __restrict__ ntiles) {
  __shared__ int s_off[N_EXPERTS];
  __shared__ int s_cur[N_EXPERTS];
  int tid = threadIdx.x;
  if (tid < N_EXPERTS) s_cur[tid] = 0;
  if (tid == 0) {
    int run = 0, tix = 0;
    for (int e = 0; e < N_EXPERTS; ++e) {
      s_off[e] = run;
      int padded = (counts[e] + (MTILE - 1)) & ~(MTILE - 1);
      int nt = padded / MTILE;
      for (int i = 0; i < nt; ++i) {
        tile_meta[2 * tix + 0] = e;
        tile_meta[2 * tix + 1] = run + i * MTILE;
        ++tix;
      }
      run += padded;
    }
    *ntiles = tix;
  }
  for (int i = tid; i < MAX_LIST; i += blockDim.x) slot_list[i] = -1;
  __syncthreads();
  for (int i = tid; i < N_SLOTS; i += blockDim.x) {
    int e = indices[i];                    // expert for slot i = token*2 + k
    int pos = atomicAdd(&s_cur[e], 1);     // ds_add_rtn_u32
    slot_list[s_off[e] + pos] = i;
  }
}

// no-return global f32 atomic add (native CDNA5 op; avoids CAS-loop lowering)
__device__ __forceinline__ void atomic_add_f32(float* p, float v) {
  asm volatile("global_atomic_add_f32 %0, %1, off" :: "v"(p), "v"(v) : "memory");
}

// ---------------------------------------------------------------------------
// Fused grouped GEMM, 32-slot tiles: X(32x1024) -> fc1 -> y*silu(z) -> fc2
// -> weighted scatter-add. 8 waves; each wave: 16 accumulator C-tiles.
// B fragments reused across both M-tiles (halves L2 weight traffic vs M=16).
// ---------------------------------------------------------------------------
__global__ __launch_bounds__(256)
void moe_gemm(const __bf16* __restrict__ fc1b,   // [E][2*D_INT][D_MODEL] bf16
              const __bf16* __restrict__ fc2b,   // [E][D_MODEL][D_INT]  bf16
              const __bf16* __restrict__ xb,     // [N_TOK][D_MODEL]     bf16
              const float*  __restrict__ wts,    // [N_TOK][TOP_K] router weights
              const int*    __restrict__ slot_list,
              const int*    __restrict__ tile_meta,
              const int*    __restrict__ ntiles,
              float* __restrict__ out) {         // [N_TOK][D_MODEL] f32 (pre-zeroed)
  __shared__ __align__(16) __bf16 sX[MTILE * D_MODEL];  // 64 KB
  __shared__ __align__(16) __bf16 sA[MTILE * D_INT];    // 32 KB
  __shared__ int   sTok[MTILE];
  __shared__ float sW[MTILE];

  int tile = blockIdx.x;
  if (tile >= *ntiles) return;                   // uniform per block
  const int e    = tile_meta[2 * tile + 0];
  const int base = tile_meta[2 * tile + 1];
  const int tid  = threadIdx.x;
  const int lane = tid & 31;
  const int wave = tid >> 5;
  const int nsel = lane >> 4;                    // 0: lanes 0-15, 1: lanes 16-31
  const int ncol = lane & 15;                    // N column / B lane column
  const int mrow = lane & 15;                    // A-matrix row within M-tile

  if (tid < MTILE) {
    int s = slot_list[base + tid];
    sTok[tid] = (s < 0) ? -1 : (s >> 1);
    sW[tid]   = (s < 0) ? 0.f : wts[s];
  }
  __syncthreads();

  // ---- stage X tile (32 rows x 1024 bf16) via async gather into LDS
  {
    const uint4* xbq = reinterpret_cast<const uint4*>(xb);  // 128 uint4 / row
    uint4* sXq = reinterpret_cast<uint4*>(sX);
    unsigned sx_base = (unsigned)(unsigned long long)(lds_char_t*)sX;
    const uint4 zz = {0u, 0u, 0u, 0u};
#pragma unroll
    for (int j = 0; j < 16; ++j) {
      int idx = tid + 256 * j;                   // 0..4095 (uint4 index)
      int r = idx >> 7, c = idx & 127;
      int t = sTok[r];
      if (t >= 0) {
        const uint4* g = xbq + (size_t)t * 128 + c;
        unsigned l = sx_base + (unsigned)idx * 16u;
        asm volatile("global_load_async_to_lds_b128 %0, %1, off"
                     :: "v"(l), "v"(g) : "memory");     // ASYNCcnt path
      } else {
        sXq[idx] = zz;                           // zero-fill padding rows
      }
    }
  }
  asm volatile("s_wait_asynccnt 0x0" ::: "memory");
  __syncthreads();

  // ---- FC1: h = X * fc1^T ; wave w owns y-cols [w*64, w*64+64) + paired z,
  //      for both 16-row M-tiles (m=0,1).
  const __bf16* w1 = fc1b + (size_t)e * (2 * D_INT) * D_MODEL;
  v8f accY[2][4] = {}, accZ[2][4] = {};
  for (int k0 = 0; k0 < D_MODEL; k0 += 32) {
    FragAB a[2];
#pragma unroll
    for (int m = 0; m < 2; ++m) {
      const uint4* sXrow =
          reinterpret_cast<const uint4*>(sX + (mrow + m * 16) * D_MODEL);
      a[m].q[0] = sXrow[(k0 + nsel * 16) >> 3];       // K[k0+16s .. +7]
      a[m].q[1] = sXrow[(k0 + nsel * 16 + 8) >> 3];   // K[k0+16s+8 .. +7]
    }
#pragma unroll
    for (int j = 0; j < 4; ++j) {
      int gy = (wave * 4 + j) * 16 + ncol;            // fc1 output row (y half)
      FragAB by, bz;
      by.v = *reinterpret_cast<const v16bf*>(w1 + (size_t)gy * D_MODEL + k0 + nsel * 16);
      bz.v = *reinterpret_cast<const v16bf*>(w1 + (size_t)(gy + D_INT) * D_MODEL + k0 + nsel * 16);
#pragma unroll
      for (int m = 0; m < 2; ++m) {
        accY[m][j] = __builtin_amdgcn_wmma_f32_16x16x32_bf16(false, a[m].v, false, by.v,
                                                             (short)0, accY[m][j], false, false);
        accZ[m][j] = __builtin_amdgcn_wmma_f32_16x16x32_bf16(false, a[m].v, false, bz.v,
                                                             (short)0, accZ[m][j], false, false);
      }
    }
  }

  // ---- gated activation a = y * silu(z), bf16 into LDS
#pragma unroll
  for (int m = 0; m < 2; ++m) {
#pragma unroll
    for (int j = 0; j < 4; ++j) {
      int p = wave * 4 + j;
#pragma unroll
      for (int r = 0; r < 8; ++r) {
        float y = accY[m][j][r], z = accZ[m][j][r];
        float av = y * z * (1.f / (1.f + __expf(-z)));
        int M = r + nsel * 8 + m * 16;
        sA[M * D_INT + p * 16 + ncol] = (__bf16)av;
      }
    }
  }
  __syncthreads();

  // ---- FC2: out_slot = a * fc2^T ; wave w owns output cols [w*128, w*128+128)
  const __bf16* w2 = fc2b + (size_t)e * D_MODEL * D_INT;
  v8f acc2[2][8] = {};
  for (int k0 = 0; k0 < D_INT; k0 += 32) {
    FragAB a[2];
#pragma unroll
    for (int m = 0; m < 2; ++m) {
      const uint4* sArow =
          reinterpret_cast<const uint4*>(sA + (mrow + m * 16) * D_INT);
      a[m].q[0] = sArow[(k0 + nsel * 16) >> 3];
      a[m].q[1] = sArow[(k0 + nsel * 16 + 8) >> 3];
    }
#pragma unroll
    for (int j = 0; j < 8; ++j) {
      int d = (wave * 8 + j) * 16 + ncol;             // fc2 output row
      FragAB b;
      b.v = *reinterpret_cast<const v16bf*>(w2 + (size_t)d * D_INT + k0 + nsel * 16);
#pragma unroll
      for (int m = 0; m < 2; ++m) {
        acc2[m][j] = __builtin_amdgcn_wmma_f32_16x16x32_bf16(false, a[m].v, false, b.v,
                                                             (short)0, acc2[m][j], false, false);
      }
    }
  }

  // ---- combine: out[t] += router_weight * slot_out  (f32 global atomics)
#pragma unroll
  for (int j = 0; j < 8; ++j) {
    int dcol = (wave * 8 + j) * 16 + ncol;
#pragma unroll
    for (int m = 0; m < 2; ++m) {
#pragma unroll
      for (int r = 0; r < 8; ++r) {
        int M = r + nsel * 8 + m * 16;
        int t = sTok[M];
        if (t >= 0)
          atomic_add_f32(out + (size_t)t * D_MODEL + dcol, acc2[m][j][r] * sW[M]);
      }
    }
  }
}

// ---------------------------------------------------------------------------
// Host launcher. Workspace layout (assumes ws_size >= ~30 MB):
//   fc1 bf16 16 MB | fc2 bf16 8 MB | x bf16 4 MB | slot list | tile meta | ntiles
// ---------------------------------------------------------------------------
extern "C" void kernel_launch(void* const* d_in, const int* in_sizes, int n_in,
                              void* d_out, int out_size, void* d_ws, size_t ws_size,
                              hipStream_t stream) {
  const float* x    = (const float*)d_in[0];
  const float* wts  = (const float*)d_in[1];
  const int*   idx  = (const int*)  d_in[2];
  const int*   cnt  = (const int*)  d_in[3];
  const float* fc1  = (const float*)d_in[4];
  const float* fc2  = (const float*)d_in[5];
  float* out = (float*)d_out;
  char*  ws  = (char*)d_ws;

  const size_t SZ_FC1 = (size_t)N_EXPERTS * 2 * D_INT * D_MODEL;  // elements
  const size_t SZ_FC2 = (size_t)N_EXPERTS * D_MODEL * D_INT;
  const size_t SZ_X   = (size_t)N_TOK * D_MODEL;
  const size_t OFF_FC1  = 0;
  const size_t OFF_FC2  = OFF_FC1 + SZ_FC1 * 2;
  const size_t OFF_XB   = OFF_FC2 + SZ_FC2 * 2;
  const size_t OFF_LIST = OFF_XB  + SZ_X * 2;
  const size_t OFF_META = OFF_LIST + (size_t)MAX_LIST * 4;
  const size_t OFF_NT   = OFF_META + (size_t)MAX_TILES * 2 * 4;

  __bf16* fc1b = (__bf16*)(ws + OFF_FC1);
  __bf16* fc2b = (__bf16*)(ws + OFF_FC2);
  __bf16* xb   = (__bf16*)(ws + OFF_XB);
  int* slot_list = (int*)(ws + OFF_LIST);
  int* tile_meta = (int*)(ws + OFF_META);
  int* ntiles    = (int*)(ws + OFF_NT);

  int n4;
  n4 = (int)(SZ_FC1 / 4);
  cvt_f32_to_bf16<<<(n4 + 255) / 256, 256, 0, stream>>>(fc1, fc1b, n4);
  n4 = (int)(SZ_FC2 / 4);
  cvt_f32_to_bf16<<<(n4 + 255) / 256, 256, 0, stream>>>(fc2, fc2b, n4);
  n4 = (int)(SZ_X / 4);
  cvt_f32_to_bf16<<<(n4 + 255) / 256, 256, 0, stream>>>(x, xb, n4);

  hipMemsetAsync(out, 0, (size_t)N_TOK * D_MODEL * sizeof(float), stream);

  build_dispatch<<<1, 256, 0, stream>>>(idx, cnt, slot_list, tile_meta, ntiles);

  moe_gemm<<<MAX_TILES, 256, 0, stream>>>(fc1b, fc2b, xb, wts, slot_list,
                                          tile_meta, ntiles, out);
}
